// NeuralRadianceField_1726576857526
// MI455X (gfx1250) — matrix operand order repaired and checked by reading, not compiled, not run on hardware
//
#include <hip/hip_runtime.h>
#include <math.h>
#include <cstdint>

// ---- WMMA vector types (gfx1250, wave32) ----
typedef __attribute__((ext_vector_type(16))) _Float16 v16h;
typedef __attribute__((ext_vector_type(8)))  _Float16 h8;
typedef __attribute__((ext_vector_type(8)))  float    v8f;

#define WAVES   4
#define TPB     (WAVES * 32)
#define PTS_BLK (WAVES * 16)
#define NTILES  32
#define WHALF   (NTILES * 32 * 16)        // 16384 f16 weight elements
#define WBYTES  (WHALF * 2)               // 32 KB pre-laid-out weight image

__device__ __forceinline__ v16h cat8(h8 lo, h8 hi) {
  return __builtin_shufflevector(lo, hi, 0,1,2,3,4,5,6,7,8,9,10,11,12,13,14,15);
}

__device__ __forceinline__ v8f wmma16x16x32(v16h a, v16h b, v8f c) {
  // D = A(16x32 f16) * B(32x16 f16) + C(16x16 f32)
  return __builtin_amdgcn_wmma_f32_16x16x32_f16(
      /*neg_a=*/false, a, /*neg_b=*/false, b,
      /*c_mod=*/(short)0, c, /*reuse_a=*/false, /*reuse_b=*/false);
}

// A-operand fetch (16x32 f16) from LDS activation row (row-major [16][Kfull]).
// ISA A-layout: lane<16 holds K {b..b+7, b+16..b+23} with b=0; lane>=16 b=8.
__device__ __forceinline__ v16h loadA(const _Float16* rowPtr, int kstep, int bOff) {
  const int base = kstep * 32 + bOff;
  h8 lo = *(const h8*)(rowPtr + base);
  h8 hi = *(const h8*)(rowPtr + base + 16);
  return cat8(lo, hi);
}

// Convert one (tile,lane) slot of the weight image: 16 f16 (K b..b+7, b+16..b+23; b=0|8 by lane half)
// values for this lane's B column.  Tile map:
//   [0..7]   layer1 wd1 32x128   (kstep 0, ntile t)
//   [8..11]  layer2 wd2 128x16   (kstep t-8)
//   [12..27] layer3 wc1 43x128 zero-padded to K=64 (ks=(t-12)>>3, nt=(t-12)&7)
//   [28..31] layer4 wc2 128x3 zero-padded to N=16 (kstep t-28)
__device__ __forceinline__ void fill_slot(int tile, int lane,
                                          const float* wd1, const float* wd2,
                                          const float* wc1, const float* wc2,
                                          _Float16* dst) {
  const int col16 = lane & 15;
  const int kbase = (lane < 16) ? 0 : 8;
  if (tile < 8) {
    const int nc = tile * 16 + col16;
    #pragma unroll
    for (int h = 0; h < 16; ++h) dst[h] = (_Float16)wd1[(kbase + h + ((h >> 3) << 3)) * 128 + nc];
  } else if (tile < 12) {
    const int ks = tile - 8;
    #pragma unroll
    for (int h = 0; h < 16; ++h) dst[h] = (_Float16)wd2[(ks * 32 + kbase + h + ((h >> 3) << 3)) * 16 + col16];
  } else if (tile < 28) {
    const int loc = tile - 12, ks = loc >> 3, nt = loc & 7;
    const int nc = nt * 16 + col16;
    #pragma unroll
    for (int h = 0; h < 16; ++h) {
      const int g = ks * 32 + kbase + h + ((h >> 3) << 3);
      dst[h] = (g < 43) ? (_Float16)wc1[g * 128 + nc] : (_Float16)0.f;
    }
  } else {
    const int ks = tile - 28;
    #pragma unroll
    for (int h = 0; h < 16; ++h) {
      const int g = ks * 32 + kbase + h + ((h >> 3) << 3);
      dst[h] = (col16 < 3) ? (_Float16)wc2[g * 3 + col16] : (_Float16)0.f;
    }
  }
}

// One-shot: weights -> f16 B-layout image in global scratch (L2-resident).
__global__ void prep_weights(const float* __restrict__ wd1, const float* __restrict__ wd2,
                             const float* __restrict__ wc1, const float* __restrict__ wc2,
                             _Float16* __restrict__ wpre) {
  const int slot = blockIdx.x * blockDim.x + threadIdx.x;   // 1024 slots
  if (slot >= NTILES * 32) return;
  __align__(16) _Float16 tmp[16];
  fill_slot(slot >> 5, slot & 31, wd1, wd2, wc1, wc2, tmp);
  h8* dst = (h8*)(wpre + slot * 16);
  dst[0] = *(const h8*)&tmp[0];
  dst[1] = *(const h8*)&tmp[8];
}

__global__ __launch_bounds__(TPB)
void nerf_fused_wmma(const float* __restrict__ coords,
                     const float* __restrict__ ray_d,
                     const float* __restrict__ grid,
                     const float* __restrict__ wd1, const float* __restrict__ bd1,
                     const float* __restrict__ wd2, const float* __restrict__ bd2,
                     const float* __restrict__ wc1, const float* __restrict__ bc1,
                     const float* __restrict__ wc2, const float* __restrict__ bc2,
                     const _Float16* __restrict__ wpre,
                     float* __restrict__ out, int Npts) {
  __shared__ __align__(32) _Float16 wB[NTILES][32][16];      // 32 KB B-tiles
  __shared__ __align__(32) _Float16 actS[WAVES][16][128];    // 16 KB act scratch
  __shared__ __align__(32) _Float16 fdirS[WAVES][16][64];    //  8 KB fdir
  __shared__ __align__(16) float    outS[WAVES][16][4];      //  1 KB out staging

  const int tid = threadIdx.x;

  // ------- Stage 0: stage pre-converted weights into LDS (async path) ------
  if (wpre != nullptr) {
    const unsigned ldsBase = (unsigned)(uintptr_t)&wB[0][0][0];  // LDS byte offset
    #pragma unroll
    for (int i = tid; i < WBYTES / 16; i += TPB) {               // 16 iters/thread
      const unsigned dst = ldsBase + (unsigned)i * 16u;
      const unsigned long long src = (unsigned long long)(uintptr_t)wpre
                                   + (unsigned long long)i * 16ull;
      asm volatile("global_load_async_to_lds_b128 %0, %1, off"
                   :: "v"(dst), "v"(src) : "memory");
    }
    asm volatile("s_wait_asynccnt 0x0" ::: "memory");
  } else {
    // Fallback: convert in-block (only if scratch was too small).
    for (int slot = tid; slot < NTILES * 32; slot += TPB)
      fill_slot(slot >> 5, slot & 31, wd1, wd2, wc1, wc2, &wB[slot >> 5][slot & 31][0]);
  }
  __syncthreads();

  const int wave   = tid >> 5;
  const int lane   = tid & 31;
  const int row16  = lane & 15;               // point-in-tile / C-col / A-row
  const int hiHalf = (lane >= 16);
  const int rowOff = hiHalf ? 8 : 0;          // C/D layout: lanes>=16 hold M=r+8
  const int bOff   = hiHalf ? 8 : 0;          // A layout K sub-offset
  const int pointBase = blockIdx.x * PTS_BLK + wave * 16;
  int p = pointBase + row16;
  if (p >= Npts) p = Npts - 1;                // clamp loads; stores guarded later

  __builtin_prefetch(&ray_d[p * 3], 0, 0);    // warm ray line for the dir embedding

  // ------- Stage 1: trilinear gather, straight into WMMA A-layout ----------
  const float3 cc = *(const float3*)(coords + (size_t)p * 3);
  const float cx = cc.x * 127.f, cy = cc.y * 127.f, cz = cc.z * 127.f;
  int ix = (int)floorf(cx); ix = ix < 0 ? 0 : (ix > 126 ? 126 : ix);
  int iy = (int)floorf(cy); iy = iy < 0 ? 0 : (iy > 126 ? 126 : iy);
  int iz = (int)floorf(cz); iz = iz < 0 ? 0 : (iz > 126 ? 126 : iz);
  const float tx = cx - (float)ix, ty = cy - (float)iy, tz = cz - (float)iz;
  const int fb = bOff;                        // this lane's feature base

  float acc[16];
  #pragma unroll
  for (int j = 0; j < 16; ++j) acc[j] = 0.f;

  #pragma unroll
  for (int c = 0; c < 8; ++c) {
    const int dx = (c >> 2) & 1, dy = (c >> 1) & 1, dz = c & 1;
    const float w = (dx ? tx : 1.f - tx) * (dy ? ty : 1.f - ty) * (dz ? tz : 1.f - tz);
    const float* gp = grid + ((size_t)((ix + dx) * 128 + (iy + dy)) * 128 + (iz + dz)) * 32 + fb;
    #pragma unroll
    for (int q = 0; q < 2; ++q) {             // feature runs [fb,fb+8), [fb+16,fb+24)
      const float4 a0 = *(const float4*)(gp + q * 16);
      const float4 a1 = *(const float4*)(gp + q * 16 + 4);
      float* d = &acc[q * 8];
      d[0] += w * a0.x; d[1] += w * a0.y; d[2] += w * a0.z; d[3] += w * a0.w;
      d[4] += w * a1.x; d[5] += w * a1.y; d[6] += w * a1.z; d[7] += w * a1.w;
    }
  }

  v16h a1;
  #pragma unroll
  for (int j = 0; j < 16; ++j) a1[j] = (_Float16)acc[j];

  // ------- Layer 1: h = relu(feats @ wd1 + bd1)  (8 WMMA) ------------------
  #pragma unroll
  for (int nt = 0; nt < 8; ++nt) {
    v16h b = *(const v16h*)&wB[nt][lane][0];
    v8f  c = {};
    c = wmma16x16x32(a1, b, c);
    const float bn = bd1[nt * 16 + row16];
    #pragma unroll
    for (int r = 0; r < 8; ++r)
      actS[wave][r + rowOff][nt * 16 + row16] = (_Float16)fmaxf(c[r] + bn, 0.f);
  }
  __syncthreads();

  // ------- Layer 2: dfeat = h @ wd2 + bd2  (4 WMMA) ------------------------
  const _Float16* actRow = &actS[wave][row16][0];
  v16h a2[4];
  #pragma unroll
  for (int ks = 0; ks < 4; ++ks) a2[ks] = loadA(actRow, ks, bOff);
  v8f c2 = {};
  #pragma unroll
  for (int ks = 0; ks < 4; ++ks)
    c2 = wmma16x16x32(a2[ks], *(const v16h*)&wB[8 + ks][lane][0], c2);

  float dfeat[8];
  const float b2 = bd2[row16];
  #pragma unroll
  for (int r = 0; r < 8; ++r) {
    dfeat[r] = c2[r] + b2;
    fdirS[wave][r + rowOff][row16] = (_Float16)dfeat[r];   // fdir cols 0..15
  }

  // ------- Dir embedding -> fdir cols 16..42 (pad 43..63 = 0) --------------
  const float3 rr = *(const float3*)(ray_d + (size_t)p * 3);
  const float rx = rr.x, ry = rr.y, rz = rr.z;
  _Float16* frow = &fdirS[wave][row16][0];
  if (!hiHalf) {
    frow[16] = (_Float16)rx; frow[17] = (_Float16)ry; frow[18] = (_Float16)rz;
    #pragma unroll
    for (int k = 0; k < 12; ++k) {
      const float d = (k % 3 == 0) ? rx : (k % 3 == 1) ? ry : rz;
      frow[19 + k] = (_Float16)__sinf(d * (float)(1 << (k / 3)));
    }
    #pragma unroll
    for (int c = 43; c < 54; ++c) frow[c] = (_Float16)0.f;
  } else {
    #pragma unroll
    for (int k = 0; k < 12; ++k) {
      const float d = (k % 3 == 0) ? rx : (k % 3 == 1) ? ry : rz;
      frow[31 + k] = (_Float16)__cosf(d * (float)(1 << (k / 3)));
    }
    #pragma unroll
    for (int c = 54; c < 64; ++c) frow[c] = (_Float16)0.f;
  }
  __syncthreads();

  // ------- Layer 3: h2 = relu(fdir @ wc1 + bc1)  (16 WMMA) -----------------
  const _Float16* fRow = &fdirS[wave][row16][0];
  v16h a3[2];
  #pragma unroll
  for (int ks = 0; ks < 2; ++ks) a3[ks] = loadA(fRow, ks, bOff);
  #pragma unroll
  for (int nt = 0; nt < 8; ++nt) {
    v8f c = {};
    #pragma unroll
    for (int ks = 0; ks < 2; ++ks)
      c = wmma16x16x32(a3[ks], *(const v16h*)&wB[12 + ks * 8 + nt][lane][0], c);
    const float bn = bc1[nt * 16 + row16];
    #pragma unroll
    for (int r = 0; r < 8; ++r)
      actS[wave][r + rowOff][nt * 16 + row16] = (_Float16)fmaxf(c[r] + bn, 0.f);
  }
  __syncthreads();

  // ------- Layer 4: rgb = sigmoid(h2 @ wc2 + bc2)  (4 WMMA) ----------------
  v16h a4[4];
  #pragma unroll
  for (int ks = 0; ks < 4; ++ks) a4[ks] = loadA(actRow, ks, bOff);
  v8f o = {};
  #pragma unroll
  for (int ks = 0; ks < 4; ++ks)
    o = wmma16x16x32(a4[ks], *(const v16h*)&wB[28 + ks][lane][0], o);

  // ------- Epilogue: stage {rgb, relu(dfeat0)} in LDS, coalesced b128 out --
  if (row16 < 3) {
    const float bn = bc2[row16];
    #pragma unroll
    for (int r = 0; r < 8; ++r) {
      const float v = o[r] + bn;
      outS[wave][r + rowOff][row16] = 1.f / (1.f + __expf(-v));
    }
  }
  if (row16 == 0) {
    #pragma unroll
    for (int r = 0; r < 8; ++r)
      outS[wave][r + rowOff][3] = fmaxf(dfeat[r], 0.f);
  }
  __syncthreads();

  if (!hiHalf) {
    const int pt = pointBase + row16;
    if (pt < Npts) {
      const float4 v = *(const float4*)&outS[wave][row16][0];
      *(float4*)(out + (size_t)pt * 4) = v;    // one global_store_b128 per point
    }
  }
}

extern "C" void kernel_launch(void* const* d_in, const int* in_sizes, int n_in,
                              void* d_out, int out_size, void* d_ws, size_t ws_size,
                              hipStream_t stream) {
  const float* coords = (const float*)d_in[0];
  const float* ray_d  = (const float*)d_in[1];
  const float* grid   = (const float*)d_in[2];
  const float* wd1    = (const float*)d_in[3];
  const float* bd1    = (const float*)d_in[4];
  const float* wd2    = (const float*)d_in[5];
  const float* bd2    = (const float*)d_in[6];
  const float* wc1    = (const float*)d_in[7];
  const float* bc1    = (const float*)d_in[8];
  const float* wc2    = (const float*)d_in[9];
  const float* bc2    = (const float*)d_in[10];

  const int Npts   = in_sizes[0] / 3;                 // 2,097,152
  const int blocks = (Npts + PTS_BLK - 1) / PTS_BLK;  // 32,768 blocks of 128 thr

  _Float16* wpre = (ws_size >= (size_t)WBYTES) ? (_Float16*)d_ws : nullptr;
  if (wpre)
    prep_weights<<<4, 256, 0, stream>>>(wd1, wd2, wc1, wc2, wpre);

  nerf_fused_wmma<<<blocks, TPB, 0, stream>>>(
      coords, ray_d, grid, wd1, bd1, wd2, bd2, wc1, bc1, wc2, bc2,
      wpre, (float*)d_out, Npts);
}
